// Head_29661044146527
// MI455X (gfx1250) — matrix operand order repaired and checked by reading, not compiled, not run on hardware
//
#include <hip/hip_runtime.h>
#include <hip/hip_bf16.h>

// Problem constants (match reference)
#define SEQ   4096
#define EMB   1024
#define HD    128
#define BATCH 2

typedef __attribute__((ext_vector_type(16))) __bf16 v16bf;
typedef __attribute__((ext_vector_type(8)))  float  v8f;

union Frag16 {
    v16bf    v;
    unsigned u[8];   // 8 dwords = 16 bf16
};

// ---- gfx1250 async global->LDS path (guarded; falls back to plain copy) ----
#ifdef __has_builtin
#  if __has_builtin(__builtin_amdgcn_global_load_async_to_lds_b32)
#    define HAVE_ASYNC_LDS 1
#  endif
#endif
#ifndef HAVE_ASYNC_LDS
#  define HAVE_ASYNC_LDS 0
#endif

#define AS_GLOBAL __attribute__((address_space(1)))
#define AS_LOCAL  __attribute__((address_space(3)))

__device__ __forceinline__ void async_copy_b32(const void* g, void* l) {
#if HAVE_ASYNC_LDS
    // builtin signature (from hipcc diagnostic): (global int*, lds int*, imm, imm)
    __builtin_amdgcn_global_load_async_to_lds_b32(
        (AS_GLOBAL int*)g, (AS_LOCAL int*)l, 0, 0);
#else
    *(unsigned*)l = *(const unsigned*)g;
#endif
}

__device__ __forceinline__ void async_drain() {
#if HAVE_ASYNC_LDS
#  if __has_builtin(__builtin_amdgcn_s_wait_asynccnt)
    __builtin_amdgcn_s_wait_asynccnt(0);
#  else
    asm volatile("s_wait_asynccnt 0x0" ::: "memory");
#  endif
#endif
}

// pack two floats into one dword of 2 x bf16 (compiler emits v_cvt_pk_bf16_f32)
__device__ __forceinline__ unsigned pk_bf16(float lo, float hi) {
    union { __bf16 h[2]; unsigned u; } r;
    r.h[0] = (__bf16)lo; r.h[1] = (__bf16)hi;
    return r.u;
}

// ---- Fragment builders ----------------------------------------------------
// 16-bit A-matrix 16x32 (MxK), wave32:
//   lanes 0-15 : M = lane,    K pairs {0,1},{2,3},{4,5},{6,7},{16,17}..{22,23}
//   lanes 16-31: M = lane-16, K pairs shifted by +8
__device__ __forceinline__ unsigned a_pair_k0(int g, int hi) {
    return (unsigned)(((g < 4) ? (g * 2) : (16 + (g - 4) * 2)) + (hi ? 8 : 0));
}

__device__ __forceinline__ v16bf load_a_frag(const __bf16* row, int hi) {
    Frag16 f;
    const unsigned* p = (const unsigned*)row;
#pragma unroll
    for (int g = 0; g < 8; ++g) f.u[g] = p[a_pair_k0(g, hi) >> 1];
    return f.v;
}

// 16-bit B-matrix 32x16 (KxN), wave32: lane&15 -> N column; elements -> K (same order as A).
__device__ __forceinline__ v16bf load_b_frag(const __bf16* col, int ld, int hi) {
    v16bf v;
#pragma unroll
    for (int s = 0; s < 16; ++s) {
        int k = (s < 8) ? (s + 8 * hi) : (16 + 8 * hi + (s - 8));
        v[s] = col[k * ld];
    }
    return v;
}

// ---- Kernel 1: QKV projection + bias, fp32 -> bf16 ------------------------
// grid = (SEQ/64, BATCH, 3), block = 256 (8 waves), double-buffered LDS.
__global__ __launch_bounds__(256) void qkv_proj_kernel(
    const float* __restrict__ X,
    const float* __restrict__ Wq, const float* __restrict__ Wk,
    const float* __restrict__ Wv,
    const float* __restrict__ bq, const float* __restrict__ bk,
    const float* __restrict__ bv,
    __bf16* __restrict__ Qo, __bf16* __restrict__ Ko, __bf16* __restrict__ Vo)
{
    const int b = blockIdx.y;
    const int z = blockIdx.z;
    const float*  W    = (z == 0) ? Wq : (z == 1) ? Wk : Wv;
    const float*  bias = (z == 0) ? bq : (z == 1) ? bk : bv;
    __bf16*       O    = (z == 0) ? Qo : (z == 1) ? Ko : Vo;

    const int q0 = blockIdx.x * 64;
    const float* Xb = X + ((size_t)b * SEQ + q0) * EMB;

    __shared__ __bf16 abuf[2][64 * 34];    // input tile 64x32, padded
    __shared__ __bf16 wbuf[2][32 * 130];   // weight tile 32x128, padded

    const int tid  = threadIdx.x;
    const int lane = tid & 31;
    const int wave = tid >> 5;
    const int hi   = lane >> 4;
    const int nl   = lane & 15;
    const int wm   = wave >> 1;          // M-tile 0..3
    const int wn   = (wave & 1) * 64;    // N half

    // Vectorized tile staging: float4 loads, packed bf16 dword stores.
    auto stage = [&](int kc, int bs) {
        {   // X tile 64x32 : thread -> (row, 8 cols)
            int r  = tid >> 2;
            int c0 = (tid & 3) * 8;
            const float4* xp = (const float4*)(Xb + (size_t)r * EMB + kc + c0);
            float4 x0 = xp[0], x1 = xp[1];
            unsigned* dst = (unsigned*)&abuf[bs][r * 34 + c0];
            dst[0] = pk_bf16(x0.x, x0.y);
            dst[1] = pk_bf16(x0.z, x0.w);
            dst[2] = pk_bf16(x1.x, x1.y);
            dst[3] = pk_bf16(x1.z, x1.w);
            if (kc + 32 < EMB)
                __builtin_prefetch(Xb + (size_t)r * EMB + kc + 32 + c0, 0, 3);
        }
        {   // W tile 32x128 : thread -> (row, 16 cols)
            int r  = tid >> 3;
            int c0 = (tid & 7) * 16;
            const float4* wp = (const float4*)(W + (size_t)(kc + r) * HD + c0);
            float4 w0 = wp[0], w1 = wp[1], w2 = wp[2], w3 = wp[3];
            unsigned* dst = (unsigned*)&wbuf[bs][r * 130 + c0];
            dst[0] = pk_bf16(w0.x, w0.y);
            dst[1] = pk_bf16(w0.z, w0.w);
            dst[2] = pk_bf16(w1.x, w1.y);
            dst[3] = pk_bf16(w1.z, w1.w);
            dst[4] = pk_bf16(w2.x, w2.y);
            dst[5] = pk_bf16(w2.z, w2.w);
            dst[6] = pk_bf16(w3.x, w3.y);
            dst[7] = pk_bf16(w3.z, w3.w);
            if (kc + 32 < EMB)
                __builtin_prefetch(W + (size_t)(kc + 32 + r) * HD + c0, 0, 3);
        }
    };

    v8f acc[4] = {};

    stage(0, 0);
    __syncthreads();
    int buf = 0;
    for (int k0 = 0; k0 < EMB; k0 += 32) {
        if (k0 + 32 < EMB) stage(k0 + 32, buf ^ 1);

        v16bf a = load_a_frag(&abuf[buf][(wm * 16 + nl) * 34], hi);
#pragma unroll
        for (int nt = 0; nt < 4; ++nt) {
            v16bf bf = load_b_frag(&wbuf[buf][wn + nt * 16 + nl], 130, hi);
            acc[nt] = __builtin_amdgcn_wmma_f32_16x16x32_bf16(
                false, a, false, bf, (short)0, acc[nt], false, false);
        }
        __syncthreads();
        buf ^= 1;
    }

    // bias + store bf16 (C layout: VGPR r -> M = r + 8*hi, lane&15 -> N)
#pragma unroll
    for (int nt = 0; nt < 4; ++nt) {
        int col = wn + nt * 16 + nl;
        float bb = bias[col];
#pragma unroll
        for (int r = 0; r < 8; ++r) {
            int row = q0 + wm * 16 + r + 8 * hi;
            O[((size_t)b * SEQ + row) * HD + col] = (__bf16)(acc[nt][r] + bb);
        }
    }
}

// ---- Kernel 2: causal flash attention -------------------------------------
// grid = (SEQ/128, BATCH), block = 256 (8 waves); wave w owns q rows
// [q0+16w, q0+16w+16). Key loop in 32-key blocks, double-buffered; V tile
// staged via async global->LDS (ASYNCcnt), K tile transposed with packed
// dword stores.
__global__ __launch_bounds__(256) void flash_attn_kernel(
    const __bf16* __restrict__ Q, const __bf16* __restrict__ K,
    const __bf16* __restrict__ V, float* __restrict__ out)
{
    const int b  = blockIdx.y;
    const int q0 = blockIdx.x * 128;
    const int tid  = threadIdx.x;
    const int lane = tid & 31;
    const int wave = tid >> 5;
    const int hi   = lane >> 4;
    const int nl   = lane & 15;

    const __bf16* Qb = Q + (size_t)b * SEQ * HD;
    const __bf16* Kb = K + (size_t)b * SEQ * HD;
    const __bf16* Vb = V + (size_t)b * SEQ * HD;

    __shared__ __bf16 kbufT[2][128 * 34];   // K^T tile: [h][key], padded
    __shared__ __bf16 vbuf[2][32 * 130];    // V tile:   [key][h], padded
    __shared__ __bf16 pbuf[8 * 16 * 34];    // per-wave P scratch (C->A relayout)

    // Stage one 32-key block: transpose K with packed pair stores; async V.
    auto stage_kv = [&](int j, int bs) {
        {   // K transpose: thread -> (2 keys, 8 h). 2 x b128 loads, 8 x b32 stores
            int kp = (tid & 15) * 2;
            int h0 = (tid >> 4) * 8;
            const uint4* r0 = (const uint4*)(Kb + (size_t)(j + kp) * HD + h0);
            const uint4* r1 = (const uint4*)(Kb + (size_t)(j + kp + 1) * HD + h0);
            uint4 a = *r0, c = *r1;
            unsigned aw[4] = {a.x, a.y, a.z, a.w};
            unsigned cw[4] = {c.x, c.y, c.z, c.w};
#pragma unroll
            for (int d = 0; d < 4; ++d) {
                unsigned lo  = (aw[d] & 0xFFFFu) | (cw[d] << 16);
                unsigned hi2 = (aw[d] >> 16) | (cw[d] & 0xFFFF0000u);
                int h = h0 + d * 2;
                *(unsigned*)&kbufT[bs][(h + 0) * 34 + kp] = lo;
                *(unsigned*)&kbufT[bs][(h + 1) * 34 + kp] = hi2;
            }
        }
        // V copy: 2048 dwords, 8 per thread, async global->LDS
#pragma unroll
        for (int c = 0; c < 8; ++c) {
            int e  = (c * 256 + tid) * 2;      // bf16 element index
            int kk = e >> 7, h = e & 127;
            async_copy_b32(Vb + (size_t)(j + kk) * HD + h,
                           &vbuf[bs][kk * 130 + h]);
        }
    };

    // Q fragments (4 K-steps over H=128), loaded once straight from global
    v16bf qa[4];
    {
        const int qrow = q0 + wave * 16 + nl;
        const unsigned* qrp = (const unsigned*)(Qb + (size_t)qrow * HD);
#pragma unroll
        for (int hs = 0; hs < 4; ++hs) {
            Frag16 f;
#pragma unroll
            for (int g = 0; g < 8; ++g)
                f.u[g] = qrp[(a_pair_k0(g, hi) + hs * 32) >> 1];
            qa[hs] = f.v;
        }
    }

    v8f acc[8] = {};
    float mrow[8], lrow[8];
#pragma unroll
    for (int r = 0; r < 8; ++r) { mrow[r] = -3.0e38f; lrow[r] = 0.0f; }

    // faithful to source bug: score *= sqrt(H); fold log2(e) for exp2
    const float sc = 11.313708498984761f * 1.4426950408889634f;

    const int jend = q0 + 128;
    stage_kv(0, 0);
    async_drain();
    __syncthreads();
    int buf = 0;

    for (int j = 0; j < jend; j += 32) {
        if (j + 32 < jend) stage_kv(j + 32, buf ^ 1);

        if (j <= q0 + wave * 16 + 15) {     // not fully above the diagonal
            // S = Q K^T : two 16x16 tiles (keys j..j+15, j+16..j+31)
            v8f s[2] = {};
#pragma unroll
            for (int t = 0; t < 2; ++t) {
#pragma unroll
                for (int hs = 0; hs < 4; ++hs) {
                    v16bf kb = load_b_frag(&kbufT[buf][(hs * 32) * 34 + t * 16 + nl], 34, hi);
                    s[t] = __builtin_amdgcn_wmma_f32_16x16x32_bf16(
                        false, qa[hs], false, kb, (short)0, s[t], false, false);
                }
            }
            // scale, causal mask, online softmax
            float mnew[8];
#pragma unroll
            for (int r = 0; r < 8; ++r) {
                int row = q0 + wave * 16 + r + 8 * hi;
#pragma unroll
                for (int t = 0; t < 2; ++t) {
                    int key = j + t * 16 + nl;
                    float vsc = s[t][r] * sc;
                    s[t][r] = (key <= row) ? vsc : -3.0e38f;
                }
                float mx = fmaxf(s[0][r], s[1][r]);
                mx = fmaxf(mx, __shfl_xor(mx, 1, 32));
                mx = fmaxf(mx, __shfl_xor(mx, 2, 32));
                mx = fmaxf(mx, __shfl_xor(mx, 4, 32));
                mx = fmaxf(mx, __shfl_xor(mx, 8, 32));
                mnew[r] = fmaxf(mrow[r], mx);
            }
#pragma unroll
            for (int r = 0; r < 8; ++r) {
                float alpha = exp2f(mrow[r] - mnew[r]);
                float p0 = exp2f(s[0][r] - mnew[r]);
                float p1 = exp2f(s[1][r] - mnew[r]);
                s[0][r] = p0; s[1][r] = p1;
                float rs = p0 + p1;
                rs += __shfl_xor(rs, 1, 32);
                rs += __shfl_xor(rs, 2, 32);
                rs += __shfl_xor(rs, 4, 32);
                rs += __shfl_xor(rs, 8, 32);
                lrow[r] = alpha * lrow[r] + rs;
                mrow[r] = mnew[r];
#pragma unroll
                for (int nt = 0; nt < 8; ++nt) acc[nt][r] *= alpha;
            }
            // P (C layout) -> LDS -> A fragment (wave-private region; LDS ops
            // are in-order within a wave). One bf16 store per element; key pairs read back as dwords.
            __bf16* pw = &pbuf[wave * 16 * 34];
#pragma unroll
            for (int r = 0; r < 8; ++r) {
                int rr = r + 8 * hi;
                pw[rr * 34 + nl]      = (__bf16)s[0][r];
                pw[rr * 34 + 16 + nl] = (__bf16)s[1][r];
            }
            Frag16 pf;
            {
                const unsigned* pr = (const unsigned*)&pw[nl * 34];
#pragma unroll
                for (int g = 0; g < 8; ++g)
                    pf.u[g] = pr[a_pair_k0(g, hi) >> 1];
            }
            // O += P V : 8 N-tiles, one 16x16x32 WMMA each
#pragma unroll
            for (int nt = 0; nt < 8; ++nt) {
                v16bf vb = load_b_frag(&vbuf[buf][nt * 16 + nl], 130, hi);
                acc[nt] = __builtin_amdgcn_wmma_f32_16x16x32_bf16(
                    false, pf.v, false, vb, (short)0, acc[nt], false, false);
            }
        }
        async_drain();
        __syncthreads();
        buf ^= 1;
    }

    // epilogue: out = acc / l
    float* ob = out + ((size_t)b * SEQ + q0 + wave * 16) * HD;
#pragma unroll
    for (int nt = 0; nt < 8; ++nt) {
        int col = nt * 16 + nl;
#pragma unroll
        for (int r = 0; r < 8; ++r) {
            int row = r + 8 * hi;
            ob[(size_t)row * HD + col] = acc[nt][r] / lrow[r];
        }
    }
}

extern "C" void kernel_launch(void* const* d_in, const int* in_sizes, int n_in,
                              void* d_out, int out_size, void* d_ws, size_t ws_size,
                              hipStream_t stream) {
    const float* input = (const float*)d_in[0];
    const float* Wq = (const float*)d_in[1];
    const float* bq = (const float*)d_in[2];
    const float* Wk = (const float*)d_in[3];
    const float* bk = (const float*)d_in[4];
    const float* Wv = (const float*)d_in[5];
    const float* bv = (const float*)d_in[6];

    // bf16 intermediates in workspace: Q | K | V  (3 * B*S*H * 2B = 6 MB)
    __bf16* qws = (__bf16*)d_ws;
    __bf16* kws = qws + (size_t)BATCH * SEQ * HD;
    __bf16* vws = kws + (size_t)BATCH * SEQ * HD;

    dim3 blk(256);
    dim3 g1(SEQ / 64, BATCH, 3);
    qkv_proj_kernel<<<g1, blk, 0, stream>>>(input, Wq, Wk, Wv, bq, bk, bv,
                                            qws, kws, vws);

    dim3 g2(SEQ / 128, BATCH);
    flash_attn_kernel<<<g2, blk, 0, stream>>>(qws, kws, vws, (float*)d_out);
}
